// CellNetMessagePassing_47459388621643
// MI455X (gfx1250) — matrix operand order repaired and checked by reading, not compiled, not run on hardware
//
#include <hip/hip_runtime.h>
#include <hip/hip_bf16.h>

#define DFEAT 256
#define AS 520   // A-tile LDS row stride (ushorts): 1040B, 1040%256==16 -> conflict-free b128
#define HS 264   // hidden LDS row stride (ushorts): 528B
#define OS 260   // output staging row stride (floats): 1040B

typedef __attribute__((ext_vector_type(16))) __bf16 v16bf;
typedef __attribute__((ext_vector_type(8)))  float  v8f;

union Frag16 { v16bf v; unsigned int u[8]; };

__device__ inline unsigned short f2bf(float x) {
    union { __bf16 h; unsigned short s; } c;
    c.h = (__bf16)x;
    return c.s;
}

__device__ inline v8f wmma_bf16(const Frag16& a, const Frag16& b, v8f c) {
    return __builtin_amdgcn_wmma_f32_16x16x32_bf16(
        /*neg_a=*/false, a.v, /*neg_b=*/false, b.v,
        /*c_mod=*/(short)0, c, /*reuse_a=*/false, /*reuse_b=*/false);
}

// ---------------------------------------------------------------------------
// Pack f32 weights (K x N row-major) into bf16 fragment-ready layout:
// for K-block kb=k/16, column n: 8 dwords (j=0..7) hold K pairs (16kb+2j, +1).
// ushort index = (((k>>4)*N + n)*8 + ((k>>1)&7))*2 + (k&1)
// => a whole B fragment half is two global_load_b128 per lane, 512B/16 lanes
//    fully coalesced.
// ---------------------------------------------------------------------------
__global__ void pack_kernel(const float* __restrict__ w,
                            unsigned short* __restrict__ o, int K, int N) {
    int i = blockIdx.x * 256 + threadIdx.x;
    if (i >= K * N) return;
    int k = i / N, n = i - k * N;
    size_t idx = (((size_t)(k >> 4) * N + n) * 8 + ((k >> 1) & 7)) * 2 + (k & 1);
    o[idx] = f2bf(w[i]);
}

__global__ void zero_kernel(float4* __restrict__ p, int n4) {
    int i = blockIdx.x * 256 + threadIdx.x;
    if (i < n4) p[i] = make_float4(0.f, 0.f, 0.f, 0.f);
}

// ---------------------------------------------------------------------------
// Edge scatter: 64 lanes per edge, each handles 4 features (float4 gather,
// 4x hardware f32 atomic add). Source features are L2-resident (<=100MB).
// ---------------------------------------------------------------------------
__global__ void agg_kernel(const float* __restrict__ feat,
                           const int* __restrict__ eidx, int Ecnt,
                           float* __restrict__ msg, float* __restrict__ deg) {
    long long t = (long long)blockIdx.x * 256 + threadIdx.x;
    int e = (int)(t >> 6);
    if (e >= Ecnt) return;
    int c = ((int)t & 63) * 4;
    int s = eidx[e];
    int d = eidx[Ecnt + e];
    float4 v = *(const float4*)(feat + (size_t)s * DFEAT + c);
    float* mp = msg + (size_t)d * DFEAT + c;
    unsafeAtomicAdd(mp + 0, v.x);
    unsafeAtomicAdd(mp + 1, v.y);
    unsafeAtomicAdd(mp + 2, v.z);
    unsafeAtomicAdd(mp + 3, v.w);
    if (c == 0) unsafeAtomicAdd(deg + d, 1.0f);
}

// ---------------------------------------------------------------------------
// Fused per-node MLP:
//   O = resid + relu([Xa | Xb*rdeg] @ W1 + b1) @ W2 + b2   (optional LN)
// 32-row tile per WG, 8 wave32. Each wave owns a 32-col strip over ALL 32
// rows (2 m-tiles x 2 n-tiles) so W1/W2 are read from L2 exactly once per WG.
// Whole A tile staged in LDS up front -> barrier-free K loops.
// ---------------------------------------------------------------------------
__global__ __launch_bounds__(256) void mlp_kernel(
    const float* __restrict__ Xa, const float* __restrict__ Xb,
    const float* __restrict__ deg,
    const unsigned short* __restrict__ W1p, const float* __restrict__ b1,
    const unsigned short* __restrict__ W2p, const float* __restrict__ b2,
    const float* __restrict__ resid,
    const float* __restrict__ lng, const float* __restrict__ lnb,
    float* __restrict__ out, int N, int do_ln) {
    // 32*AS*2 = 33280 (A tile, bf16)  ALIASED with 32*OS*4 = 33280 (out stage)
    // 32*HS*2 = 16896 (hidden, bf16)
    // 2048 (reduction) + 256 (mu/rsig)
    __shared__ __align__(16) unsigned char smem[33280 + 16896 + 2048 + 256];
    unsigned short* a_sh = (unsigned short*)smem;
    float*          o_sh = (float*)smem;                       // alias of a_sh
    unsigned short* h_sh = (unsigned short*)(smem + 33280);
    float*          red_s = (float*)(smem + 33280 + 16896);
    float*          red_q = red_s + 256;
    float*          mu_sh = (float*)(smem + 33280 + 16896 + 2048);
    float*          rs_sh = mu_sh + 32;

    const int tid  = threadIdx.x;
    const int lane = tid & 31;
    const int wave = tid >> 5;    // 0..7 : 32-col strip
    const int hh   = lane >> 4;   // lane half
    const int l16  = lane & 15;
    const int rowbase = blockIdx.x * 32;

    // ---- stage entire A tile (32 x 512, bf16) in LDS, degree-normalized ----
    const int lrow = tid >> 3;
    const int lcol = (tid & 7) * 4;
    const int grow = min(rowbase + lrow, N - 1);
    const float rdeg = 1.0f / fmaxf(deg[grow], 1.0f);
    const float* xa = Xa + (size_t)grow * DFEAT;
    const float* xb = Xb + (size_t)grow * DFEAT;
#pragma unroll
    for (int cc = 0; cc < 16; cc++) {
        const int col = cc * 32 + lcol;
        float4 f;
        float sc;
        if (col < DFEAT) { f = *(const float4*)(xa + col); sc = 1.0f; }
        else             { f = *(const float4*)(xb + col - DFEAT); sc = rdeg; }
        union { unsigned short s[4]; uint2 d; } pk;
        pk.s[0] = f2bf(f.x * sc);
        pk.s[1] = f2bf(f.y * sc);
        pk.s[2] = f2bf(f.z * sc);
        pk.s[3] = f2bf(f.w * sc);
        *(uint2*)&a_sh[lrow * AS + col] = pk.d;
    }

    float b1v[2], b2v[2];
#pragma unroll
    for (int nt = 0; nt < 2; nt++) {
        int n = wave * 32 + nt * 16 + l16;
        b1v[nt] = b1[n];
        b2v[nt] = b2[n];
    }

    const v8f vzero = {0.f, 0.f, 0.f, 0.f, 0.f, 0.f, 0.f, 0.f};
    v8f acc[2][2];
#pragma unroll
    for (int mt = 0; mt < 2; mt++)
#pragma unroll
        for (int nt = 0; nt < 2; nt++) acc[mt][nt] = vzero;

    __syncthreads();

    // ================= GEMM1: hidden = relu(X @ W1 + b1), K = 512 ==========
    for (int kk = 0; kk < 2 * DFEAT; kk += 32) {
        Frag16 A[2];
#pragma unroll
        for (int mt = 0; mt < 2; mt++) {
            const int arow = mt * 16 + l16;
            uint4 q0 = *(const uint4*)&a_sh[arow * AS + kk + hh * 16];
            uint4 q1 = *(const uint4*)&a_sh[arow * AS + kk + hh * 16 + 8];
            A[mt].u[0] = q0.x; A[mt].u[1] = q0.y; A[mt].u[2] = q0.z; A[mt].u[3] = q0.w;
            A[mt].u[4] = q1.x; A[mt].u[5] = q1.y; A[mt].u[6] = q1.z; A[mt].u[7] = q1.w;
        }
        Frag16 B[2];
        const unsigned int* wq = (const unsigned int*)W1p +
            ((size_t)((kk >> 4) + hh) * DFEAT + wave * 32 + l16) * 8;
#pragma unroll
        for (int nt = 0; nt < 2; nt++) {
            uint4 q0 = *(const uint4*)(wq + nt * 128);      // +16 cols = 128 dw
            uint4 q1 = *(const uint4*)(wq + nt * 128 + 4);
            B[nt].u[0] = q0.x; B[nt].u[1] = q0.y; B[nt].u[2] = q0.z; B[nt].u[3] = q0.w;
            B[nt].u[4] = q1.x; B[nt].u[5] = q1.y; B[nt].u[6] = q1.z; B[nt].u[7] = q1.w;
        }
#pragma unroll
        for (int mt = 0; mt < 2; mt++)
#pragma unroll
            for (int nt = 0; nt < 2; nt++)
                acc[mt][nt] = wmma_bf16(A[mt], B[nt], acc[mt][nt]);
    }

    // bias + relu -> LDS hidden (bf16). C layout: VGPR c -> row c+8*hh, col l16.
#pragma unroll
    for (int mt = 0; mt < 2; mt++)
#pragma unroll
        for (int nt = 0; nt < 2; nt++) {
            const int n = wave * 32 + nt * 16 + l16;
#pragma unroll
            for (int c = 0; c < 8; c++) {
                float v = fmaxf(acc[mt][nt][c] + b1v[nt], 0.0f);
                h_sh[(mt * 16 + c + 8 * hh) * HS + n] = f2bf(v);
            }
        }
    __syncthreads();   // hidden complete; a_sh region now dead (reused as o_sh)

    // ================= GEMM2: O = H @ W2 + b2, K = 256 =====================
    v8f acc2[2][2];
#pragma unroll
    for (int mt = 0; mt < 2; mt++)
#pragma unroll
        for (int nt = 0; nt < 2; nt++) acc2[mt][nt] = vzero;
    for (int kk = 0; kk < DFEAT; kk += 32) {
        Frag16 A[2];
#pragma unroll
        for (int mt = 0; mt < 2; mt++) {
            const int arow = mt * 16 + l16;
            uint4 q0 = *(const uint4*)&h_sh[arow * HS + kk + hh * 16];
            uint4 q1 = *(const uint4*)&h_sh[arow * HS + kk + hh * 16 + 8];
            A[mt].u[0] = q0.x; A[mt].u[1] = q0.y; A[mt].u[2] = q0.z; A[mt].u[3] = q0.w;
            A[mt].u[4] = q1.x; A[mt].u[5] = q1.y; A[mt].u[6] = q1.z; A[mt].u[7] = q1.w;
        }
        Frag16 B[2];
        const unsigned int* wq = (const unsigned int*)W2p +
            ((size_t)((kk >> 4) + hh) * DFEAT + wave * 32 + l16) * 8;
#pragma unroll
        for (int nt = 0; nt < 2; nt++) {
            uint4 q0 = *(const uint4*)(wq + nt * 128);
            uint4 q1 = *(const uint4*)(wq + nt * 128 + 4);
            B[nt].u[0] = q0.x; B[nt].u[1] = q0.y; B[nt].u[2] = q0.z; B[nt].u[3] = q0.w;
            B[nt].u[4] = q1.x; B[nt].u[5] = q1.y; B[nt].u[6] = q1.z; B[nt].u[7] = q1.w;
        }
#pragma unroll
        for (int mt = 0; mt < 2; mt++)
#pragma unroll
            for (int nt = 0; nt < 2; nt++)
                acc2[mt][nt] = wmma_bf16(A[mt], B[nt], acc2[mt][nt]);
    }

    // epilogue (pre-residual) to LDS f32 (aliases dead A tile)
    __syncthreads();   // ensure all GEMM2 h_sh reads done before o_sh writes land
#pragma unroll
    for (int mt = 0; mt < 2; mt++)
#pragma unroll
        for (int nt = 0; nt < 2; nt++) {
            const int n = wave * 32 + nt * 16 + l16;
#pragma unroll
            for (int c = 0; c < 8; c++)
                o_sh[(mt * 16 + c + 8 * hh) * OS + n] = acc2[mt][nt][c] + b2v[nt];
        }
    __syncthreads();

    // ============== residual add + (optional) LayerNorm + store ============
    const int orow = tid >> 3;          // 8 threads per row, 32 cols each
    const int ocol = (tid & 7) * 32;
    const int growO = rowbase + orow;
    const float* rp = resid + (size_t)min(growO, N - 1) * DFEAT + ocol;
    float vals[32];
    float s = 0.f, q = 0.f;
#pragma unroll
    for (int i = 0; i < 8; i++) {
        float4 r = *(const float4*)(rp + i * 4);
        float4 o = *(const float4*)&o_sh[orow * OS + ocol + i * 4];
        float v0 = o.x + r.x, v1 = o.y + r.y, v2 = o.z + r.z, v3 = o.w + r.w;
        vals[i * 4 + 0] = v0; vals[i * 4 + 1] = v1;
        vals[i * 4 + 2] = v2; vals[i * 4 + 3] = v3;
        s += v0 + v1 + v2 + v3;
        q += v0 * v0 + v1 * v1 + v2 * v2 + v3 * v3;
    }
    red_s[tid] = s;
    red_q[tid] = q;
    __syncthreads();
    if ((tid & 7) == 0) {
        float S = 0.f, Q = 0.f;
        for (int j = 0; j < 8; j++) { S += red_s[tid + j]; Q += red_q[tid + j]; }
        float mu = S * (1.0f / DFEAT);
        float var = Q * (1.0f / DFEAT) - mu * mu;
        mu_sh[orow] = mu;
        rs_sh[orow] = rsqrtf(var + 1e-5f);
    }
    __syncthreads();
    if (growO < N) {
        float* op = out + (size_t)growO * DFEAT + ocol;
        if (do_ln) {
            const float mu = mu_sh[orow], rs = rs_sh[orow];
#pragma unroll
            for (int i = 0; i < 8; i++) {
                const int n = ocol + i * 4;
                float4 w;
                w.x = (vals[i * 4 + 0] - mu) * rs * lng[n + 0] + lnb[n + 0];
                w.y = (vals[i * 4 + 1] - mu) * rs * lng[n + 1] + lnb[n + 1];
                w.z = (vals[i * 4 + 2] - mu) * rs * lng[n + 2] + lnb[n + 2];
                w.w = (vals[i * 4 + 3] - mu) * rs * lng[n + 3] + lnb[n + 3];
                *(float4*)(op + i * 4) = w;
            }
        } else {
#pragma unroll
            for (int i = 0; i < 8; i++) {
                float4 w;
                w.x = vals[i * 4 + 0]; w.y = vals[i * 4 + 1];
                w.z = vals[i * 4 + 2]; w.w = vals[i * 4 + 3];
                *(float4*)(op + i * 4) = w;
            }
        }
    }
}

// ---------------------------------------------------------------------------
extern "C" void kernel_launch(void* const* d_in, const int* in_sizes, int n_in,
                              void* d_out, int out_size, void* d_ws,
                              size_t ws_size, hipStream_t stream) {
    (void)n_in; (void)out_size; (void)ws_size;

    const float* cell_h = (const float*)d_in[0];
    const float* net_h  = (const float*)d_in[1];
    const int* e_c2n = (const int*)d_in[2];
    const int* e_n2c = (const int*)d_in[3];
    const int* e_c2c = (const int*)d_in[4];
    const float* w1[3]  = {(const float*)d_in[5], (const float*)d_in[9],  (const float*)d_in[13]};
    const float* bb1[3] = {(const float*)d_in[6], (const float*)d_in[10], (const float*)d_in[14]};
    const float* w2[3]  = {(const float*)d_in[7], (const float*)d_in[11], (const float*)d_in[15]};
    const float* bb2[3] = {(const float*)d_in[8], (const float*)d_in[12], (const float*)d_in[16]};
    const float* net_ln_g  = (const float*)d_in[17];
    const float* net_ln_b  = (const float*)d_in[18];
    const float* cell_ln_g = (const float*)d_in[19];
    const float* cell_ln_b = (const float*)d_in[20];

    const int NCELL = in_sizes[0] / DFEAT;
    const int NNET  = in_sizes[1] / DFEAT;
    const int E1 = in_sizes[2] / 2;
    const int E2 = in_sizes[3] / 2;
    const int E3 = in_sizes[4] / 2;

    // workspace layout
    float* msg = (float*)d_ws;                                 // NCELL*256 f32
    float* deg = msg + (size_t)NCELL * DFEAT;                  // NCELL (+pad)
    float* cell2 = deg + (((size_t)NCELL + 15) & ~(size_t)15); // NCELL*256 f32
    unsigned short* wp = (unsigned short*)(cell2 + (size_t)NCELL * DFEAT);
    unsigned short* w1p[3];
    unsigned short* w2p[3];
    for (int i = 0; i < 3; i++) {
        w1p[i] = wp; wp += (size_t)2 * DFEAT * DFEAT;  // 512*256 bf16
        w2p[i] = wp; wp += (size_t)DFEAT * DFEAT;      // 256*256 bf16
    }

    float* outCell = (float*)d_out;
    float* outNet  = outCell + (size_t)NCELL * DFEAT;

    // pack all weights to bf16 (fragment-ready layout)
    for (int i = 0; i < 3; i++) {
        pack_kernel<<<(2 * DFEAT * DFEAT + 255) / 256, 256, 0, stream>>>(
            w1[i], w1p[i], 2 * DFEAT, DFEAT);
        pack_kernel<<<(DFEAT * DFEAT + 255) / 256, 256, 0, stream>>>(
            w2[i], w2p[i], DFEAT, DFEAT);
    }

    auto agg_step = [&](const float* feat, const int* eidx, int Ecnt, int Ndst) {
        int n4 = Ndst * DFEAT / 4;
        zero_kernel<<<(n4 + 255) / 256, 256, 0, stream>>>((float4*)msg, n4);
        int d4 = (Ndst + 3) / 4;
        zero_kernel<<<(d4 + 255) / 256, 256, 0, stream>>>((float4*)deg, d4);
        unsigned int g = (unsigned int)(((long long)Ecnt * 64 + 255) / 256);
        agg_kernel<<<g, 256, 0, stream>>>(feat, eidx, Ecnt, msg, deg);
    };

    // ---- round 1: cells -> nets, LN(net)
    agg_step(cell_h, e_c2n, E1, NNET);
    mlp_kernel<<<(NNET + 31) / 32, 256, 0, stream>>>(
        net_h, msg, deg, w1p[0], bb1[0], w2p[0], bb2[0], net_h,
        net_ln_g, net_ln_b, outNet, NNET, 1);

    // ---- round 2: nets -> cells, no LN
    agg_step(outNet, e_n2c, E2, NCELL);
    mlp_kernel<<<(NCELL + 31) / 32, 256, 0, stream>>>(
        cell_h, msg, deg, w1p[1], bb1[1], w2p[1], bb2[1], cell_h,
        cell_ln_g, cell_ln_b, cell2, NCELL, 0);

    // ---- round 3: cells -> cells, LN(cell)
    agg_step(cell2, e_c2c, E3, NCELL);
    mlp_kernel<<<(NCELL + 31) / 32, 256, 0, stream>>>(
        cell2, msg, deg, w1p[2], bb1[2], w2p[2], bb2[2], cell2,
        cell_ln_g, cell_ln_b, outCell, NCELL, 1);
}
